// TransformerBlock_48120813584843
// MI455X (gfx1250) — compile-verified, matrix operand order checked by reading, not run on hardware
//
#include <hip/hip_runtime.h>

#define DEVI __device__ __forceinline__

typedef __attribute__((ext_vector_type(16))) __bf16 v16bf;
typedef __attribute__((ext_vector_type(8)))  float  v8f;

union FragU { unsigned u[8]; v16bf v; };

// ---- CDNA5 async global->LDS support (toolchain-guarded) -------------------
#define AS1 __attribute__((address_space(1)))
#define AS3 __attribute__((address_space(3)))

// clang's builtin prototype uses gcc-style vector_size(16) int ("V4i")
typedef int v4i_b128 __attribute__((vector_size(16)));

#if __has_builtin(__builtin_amdgcn_global_load_async_to_lds_b128)
#define HAVE_ASYNC_LDS 1
#else
#define HAVE_ASYNC_LDS 0
#endif

DEVI AS3 v4i_b128* lds_cast(void* p) {
  // generic LDS pointer: low 32 bits are the LDS byte offset (aperture scheme)
  return (AS3 v4i_b128*)(unsigned)(unsigned long long)p;
}
DEVI AS1 v4i_b128* gbl_cast(const void* p) {
  return (AS1 v4i_b128*)(unsigned long long)p;
}

DEVI void wait_async0() {
#if HAVE_ASYNC_LDS
#if __has_builtin(__builtin_amdgcn_s_wait_asynccnt)
  __builtin_amdgcn_s_wait_asynccnt(0);
#else
  asm volatile("s_wait_asynccnt 0" ::: "memory");
#endif
#endif
}

// fp32 -> bf16, round-to-nearest-even
DEVI unsigned short f2bf(float f) {
  unsigned x = __float_as_uint(f);
  unsigned r = x + 0x7FFFu + ((x >> 16) & 1u);
  return (unsigned short)(r >> 16);
}

// dword offset (k>>1) of fragment dword j for lane-half h within a 32-wide K slice.
// 16-bit A layout: lanes 0-15: K=0..7 then 16..23; lanes 16-31: K=8..15 then 24..31.
// B uses the mirrored convention (pairs along K per dword, lane = column).
DEVI int kpair_off(int j, int h) {
  return ((j < 4) ? 0 : 8) + (j & 3) + h * 4;
}

DEVI v8f wmma_bf16(const FragU& a, const FragU& b, v8f c) {
  return __builtin_amdgcn_wmma_f32_16x16x32_bf16(false, a.v, false, b.v, (short)0, c,
                                                 false, false);
}

// ---------------------------------------------------------------------------
// fp32 -> bf16 cast (2 elements / thread, packed dword stores)
// ---------------------------------------------------------------------------
__global__ __launch_bounds__(256)
void cast_f32_bf16(const float* __restrict__ in, unsigned* __restrict__ out, long n2) {
  long i = (long)blockIdx.x * blockDim.x + threadIdx.x;
  if (i < n2) {
    float a = in[2 * i], b = in[2 * i + 1];
    out[i] = (unsigned)f2bf(a) | ((unsigned)f2bf(b) << 16);
  }
}

// ---------------------------------------------------------------------------
// bf16 WMMA GEMM: C[M,N] = A[M,K] * B[K,N], A/B bf16 row-major, f32 accum.
// Block tile 128x128x32, 8 waves (4x2), wave tile 32x64 = 2x4 16x16 frags.
// Double-buffered LDS; A tile staged via GLOBAL_LOAD_ASYNC_TO_LDS_B128 when
// available (ASYNCcnt), B tile K-pair-repacked through VGPRs (loads issued one
// iteration ahead). One barrier per K-step.
// EPI: 0 = bf16 Cb | 1 = relu->bf16 Cb | 2 = +resid, f32 Cf AND bf16 Cb
//      3 = +resid, f32 Cf only
// ---------------------------------------------------------------------------
template <int EPI>
__global__ __launch_bounds__(256)
void gemm_bf16(const unsigned short* __restrict__ A,
               const unsigned short* __restrict__ B,
               const float* __restrict__ resid,
               float* __restrict__ Cf,
               unsigned short* __restrict__ Cb,
               int M, int N, int K) {
  constexpr int BM = 128, BN = 128, BK = 32;
  constexpr int APITCH = 17;   // dwords per A row (16 K-pairs + 1 pad)
  constexpr int BPITCH = 132;  // dwords per packed-B k2 row (128 cols + 4 pad)
  constexpr int ASZ = BM * APITCH;
  constexpr int BSZ = (BK / 2) * BPITCH;
  __shared__ unsigned lds_a[2 * ASZ];
  __shared__ unsigned lds_b[2 * BSZ];

  const int tid  = threadIdx.x;
  const int wave = tid >> 5, lane = tid & 31;
  const int wm   = wave >> 1, wn = wave & 1;
  const int half = lane >> 4, ln = lane & 15;

  const long bm0 = (long)blockIdx.y * BM;
  const long bn0 = (long)blockIdx.x * BN;

  v8f acc[2][4] = {};

  // staging indices: A: 2 threads/row x 128 rows; B: 8 threads/row x 32 rows
  const int ar = tid >> 1;
  const int ac = (tid & 1) * 16;
  const int br = tid >> 3;
  const int bc = (tid & 7) * 16;

  const unsigned short* Arow = A + (bm0 + ar) * (long)K + ac;
  const unsigned short* Brow = B + (long)br * N + bn0 + bc;

  uint4 breg0, breg1;

  auto loadA = [&](int k0, int buf) {
    const unsigned short* src = Arow + k0;
    unsigned* dst = &lds_a[buf * ASZ + ar * APITCH + (ac >> 1)];
#if HAVE_ASYNC_LDS
    __builtin_amdgcn_global_load_async_to_lds_b128(gbl_cast(src), lds_cast(dst), 0, 0);
    __builtin_amdgcn_global_load_async_to_lds_b128(gbl_cast(src + 8), lds_cast(dst + 4), 0, 0);
#else
    const uint4* ag = (const uint4*)src;
    uint4 a0 = ag[0], a1 = ag[1];
    dst[0] = a0.x; dst[1] = a0.y; dst[2] = a0.z; dst[3] = a0.w;
    dst[4] = a1.x; dst[5] = a1.y; dst[6] = a1.z; dst[7] = a1.w;
#endif
  };
  auto loadB = [&](int k0) {
    const uint4* bg = (const uint4*)(Brow + (long)k0 * N);
    breg0 = bg[0];
    breg1 = bg[1];
  };
  auto storeB = [&](int buf) {
    unsigned bw[8] = {breg0.x, breg0.y, breg0.z, breg0.w, breg1.x, breg1.y, breg1.z, breg1.w};
    const int k2 = br >> 1, hb = br & 1;
    unsigned short* lb = (unsigned short*)&lds_b[buf * BSZ];
#pragma unroll
    for (int e = 0; e < 8; ++e) {
      unsigned w = bw[e];
      int col = bc + e * 2;
      lb[(unsigned)(k2 * BPITCH + col) * 2 + hb]     = (unsigned short)(w & 0xFFFFu);
      lb[(unsigned)(k2 * BPITCH + col + 1) * 2 + hb] = (unsigned short)(w >> 16);
    }
  };

  const int nk = K / BK;
  loadA(0, 0);
  loadB(0);
  storeB(0);
  wait_async0();
  __syncthreads();

  for (int kt = 0; kt < nk; ++kt) {
    const int cur = kt & 1, nxt = cur ^ 1;
    const bool more = (kt + 1) < nk;
    if (more) {
      loadA((kt + 1) * BK, nxt);  // async copy overlaps with the WMMAs below
      loadB((kt + 1) * BK);
    }

    FragU af[2], bfr[4];
#pragma unroll
    for (int mf = 0; mf < 2; ++mf) {
      const unsigned* src = &lds_a[cur * ASZ + (wm * 32 + mf * 16 + ln) * APITCH];
#pragma unroll
      for (int j = 0; j < 8; ++j) af[mf].u[j] = src[kpair_off(j, half)];
    }
#pragma unroll
    for (int nf = 0; nf < 4; ++nf) {
      const int cb = wn * 64 + nf * 16 + ln;
      const unsigned* src = &lds_b[cur * BSZ];
#pragma unroll
      for (int j = 0; j < 8; ++j) bfr[nf].u[j] = src[kpair_off(j, half) * BPITCH + cb];
    }
#pragma unroll
    for (int mf = 0; mf < 2; ++mf)
#pragma unroll
      for (int nf = 0; nf < 4; ++nf) acc[mf][nf] = wmma_bf16(af[mf], bfr[nf], acc[mf][nf]);

    if (more) {
      storeB(nxt);
      wait_async0();
      __syncthreads();
    }
  }

  // epilogue: accumulator layout: VGPR i -> row (i + half*8), col = ln
#pragma unroll
  for (int mf = 0; mf < 2; ++mf) {
    const long row0 = bm0 + wm * 32 + mf * 16 + half * 8;
#pragma unroll
    for (int nf = 0; nf < 4; ++nf) {
      const long col = bn0 + wn * 64 + nf * 16 + ln;
#pragma unroll
      for (int i = 0; i < 8; ++i) {
        long idx = (row0 + i) * (long)N + col;
        float v = acc[mf][nf][i];
        if (EPI == 1) v = v > 0.f ? v : 0.f;
        if (EPI == 2 || EPI == 3) v += resid[idx];
        if (EPI == 2 || EPI == 3) Cf[idx] = v;
        if (EPI == 0 || EPI == 1 || EPI == 2) Cb[idx] = f2bf(v);
      }
    }
  }
}

// ---------------------------------------------------------------------------
// 16x16 score tile = q_tile @ k_tile^T over D, software-pipelined (ping-pong
// fragment registers so global loads overlap the WMMAs).
// ---------------------------------------------------------------------------
DEVI v8f score_tile(const unsigned* __restrict__ qrow, const unsigned* __restrict__ krow,
                    int D, int half) {
  v8f s = {};
  FragU a0, b0, a1, b1;
#pragma unroll
  for (int j = 0; j < 8; ++j) {
    int o = kpair_off(j, half);
    a0.u[j] = qrow[o];
    b0.u[j] = krow[o];
  }
  for (int d0 = 0; d0 < D; d0 += 64) {
    const int o1 = (d0 + 32) >> 1;
#pragma unroll
    for (int j = 0; j < 8; ++j) {
      int o = o1 + kpair_off(j, half);
      a1.u[j] = qrow[o];
      b1.u[j] = krow[o];
    }
    s = wmma_bf16(a0, b0, s);
    if (d0 + 64 < D) {
      const int o2 = (d0 + 64) >> 1;
#pragma unroll
      for (int j = 0; j < 8; ++j) {
        int o = o2 + kpair_off(j, half);
        a0.u[j] = qrow[o];
        b0.u[j] = krow[o];
      }
    }
    s = wmma_bf16(a1, b1, s);
  }
  return s;
}

// ---------------------------------------------------------------------------
// Causal softmax probability matrix via WMMA q@k^T, two-pass online softmax.
// One wave per 16-query tile. B-frags (= k^T) read straight from row-major k:
// K-dim pairs are contiguous in memory, so frag dwords are plain b32 loads.
// Writes P[b, s, t] (bf16), zero above the diagonal.
// ---------------------------------------------------------------------------
__global__ __launch_bounds__(128)
void attn_pmat(const unsigned short* __restrict__ Q,
               const unsigned short* __restrict__ Km,
               unsigned short* __restrict__ P,
               int S, int D, float scale) {
  const int wave = threadIdx.x >> 5, lane = threadIdx.x & 31;
  const int half = lane >> 4, ln = lane & 15;
  const int b  = blockIdx.y;
  const int qt = blockIdx.x * (blockDim.x >> 5) + wave;
  const int q0 = qt * 16;

  const unsigned* qrow  = (const unsigned*)(Q + (long)b * S * D) + (((long)(q0 + ln) * D) >> 1);
  const unsigned* kbase = (const unsigned*)(Km + (long)b * S * D);

  float m[8], l[8];
#pragma unroll
  for (int i = 0; i < 8; ++i) { m[i] = -1e30f; l[i] = 0.f; }

  const int nkt = qt + 1;

  // ---- pass 1: row max + normalizer (online) ----
  for (int kt = 0; kt < nkt; ++kt) {
    const int k0 = kt * 16;
    const unsigned* krow = kbase + (((long)(k0 + ln) * D) >> 1);
    if (kt + 1 < nkt)  // speculative prefetch of next key tile row
      __builtin_prefetch(kbase + (((long)(k0 + 16 + ln) * D) >> 1), 0, 1);
    v8f s = score_tile(qrow, krow, D, half);
    const bool diag = (kt == qt);
#pragma unroll
    for (int i = 0; i < 8; ++i) {
      const int qi = q0 + i + half * 8;
      float sv = s[i] * scale;
      if (diag && (k0 + ln) > qi) sv = -1e30f;
      float t = sv;
      t = fmaxf(t, __shfl_xor(t, 1));
      t = fmaxf(t, __shfl_xor(t, 2));
      t = fmaxf(t, __shfl_xor(t, 4));
      t = fmaxf(t, __shfl_xor(t, 8));
      const float mn = fmaxf(m[i], t);
      float p = __expf(sv - mn);
      p += __shfl_xor(p, 1);
      p += __shfl_xor(p, 2);
      p += __shfl_xor(p, 4);
      p += __shfl_xor(p, 8);
      l[i] = l[i] * __expf(m[i] - mn) + p;
      m[i] = mn;
    }
  }
  float inv_l[8];
#pragma unroll
  for (int i = 0; i < 8; ++i) inv_l[i] = 1.f / l[i];

  // ---- pass 2: recompute scores, write normalized probs ----
  unsigned short* pb = P + (long)b * S * S;
  const int nt = S >> 4;
  for (int kt = 0; kt < nt; ++kt) {
    const int k0 = kt * 16;
    if (kt < nkt) {
      const unsigned* krow = kbase + (((long)(k0 + ln) * D) >> 1);
      v8f s = score_tile(qrow, krow, D, half);
      const bool diag = (kt == qt);
#pragma unroll
      for (int i = 0; i < 8; ++i) {
        const int qi = q0 + i + half * 8;
        float pv = 0.f;
        if (!(diag && (k0 + ln) > qi)) pv = __expf(s[i] * scale - m[i]) * inv_l[i];
        pb[(long)qi * S + k0 + ln] = f2bf(pv);
      }
    } else {
#pragma unroll
      for (int i = 0; i < 8; ++i) pb[(long)(q0 + i + half * 8) * S + k0 + ln] = 0;
    }
  }
}

// ---------------------------------------------------------------------------
// Host orchestration
// ---------------------------------------------------------------------------
extern "C" void kernel_launch(void* const* d_in, const int* in_sizes, int n_in,
                              void* d_out, int out_size, void* d_ws, size_t ws_size,
                              hipStream_t stream) {
  (void)in_sizes; (void)n_in; (void)out_size; (void)ws_size;
  const int B = 4, S = 2048, D = 1024, F = 4096;
  const int M = B * S;  // 8192

  const float* x   = (const float*)d_in[0];
  const float* Wq  = (const float*)d_in[1];
  const float* Wk  = (const float*)d_in[2];
  const float* Wv  = (const float*)d_in[3];
  const float* Wfc = (const float*)d_in[4];
  const float* Wfo = (const float*)d_in[5];

  char* wsp = (char*)d_ws;
  auto alloc = [&](size_t bytes) -> char* {
    char* p = wsp;
    wsp += (bytes + 255) & ~(size_t)255;
    return p;
  };
  unsigned short* x_bf   = (unsigned short*)alloc((size_t)M * D * 2);
  unsigned short* Wq_bf  = (unsigned short*)alloc((size_t)D * D * 2);
  unsigned short* Wk_bf  = (unsigned short*)alloc((size_t)D * D * 2);
  unsigned short* Wv_bf  = (unsigned short*)alloc((size_t)D * D * 2);
  unsigned short* Wfc_bf = (unsigned short*)alloc((size_t)D * F * 2);
  unsigned short* Wfo_bf = (unsigned short*)alloc((size_t)F * D * 2);
  unsigned short* q_bf   = (unsigned short*)alloc((size_t)M * D * 2);
  unsigned short* k_bf   = (unsigned short*)alloc((size_t)M * D * 2);
  unsigned short* v_bf   = (unsigned short*)alloc((size_t)M * D * 2);
  unsigned short* P      = (unsigned short*)alloc((size_t)B * S * S * 2);
  float*          h_f    = (float*)alloc((size_t)M * D * 4);
  unsigned short* h_bf   = (unsigned short*)alloc((size_t)M * D * 2);
  unsigned short* act_bf = (unsigned short*)alloc((size_t)M * F * 2);

  auto cast = [&](const float* in, unsigned short* out, long n) {
    long n2 = n / 2;
    unsigned blocks = (unsigned)((n2 + 255) / 256);
    cast_f32_bf16<<<dim3(blocks), dim3(256), 0, stream>>>(in, (unsigned*)out, n2);
  };
  cast(x,   x_bf,   (long)M * D);
  cast(Wq,  Wq_bf,  (long)D * D);
  cast(Wk,  Wk_bf,  (long)D * D);
  cast(Wv,  Wv_bf,  (long)D * D);
  cast(Wfc, Wfc_bf, (long)D * F);
  cast(Wfo, Wfo_bf, (long)F * D);

  const dim3 blk(256);
  // QKV projections: [8192,1024] x [1024,1024]
  gemm_bf16<0><<<dim3(D / 128, M / 128), blk, 0, stream>>>(x_bf, Wq_bf, nullptr, nullptr,
                                                           q_bf, M, D, D);
  gemm_bf16<0><<<dim3(D / 128, M / 128), blk, 0, stream>>>(x_bf, Wk_bf, nullptr, nullptr,
                                                           k_bf, M, D, D);
  gemm_bf16<0><<<dim3(D / 128, M / 128), blk, 0, stream>>>(x_bf, Wv_bf, nullptr, nullptr,
                                                           v_bf, M, D, D);

  // Causal softmax probabilities (scale = 1/sqrt(1024) = 1/32)
  attn_pmat<<<dim3(S / 16 / 4, B), dim3(128), 0, stream>>>(q_bf, k_bf, P, S, D, 0.03125f);

  // h = x + P @ v  (per batch): [2048,2048] x [2048,1024]
  for (int b = 0; b < B; ++b) {
    gemm_bf16<2><<<dim3(D / 128, S / 128), blk, 0, stream>>>(
        P + (long)b * S * S, v_bf + (long)b * S * D, x + (long)b * S * D,
        h_f + (long)b * S * D, h_bf + (long)b * S * D, S, D, S);
  }

  // MLP up: relu(h @ Wfc) -> bf16  [8192,1024] x [1024,4096]
  gemm_bf16<1><<<dim3(F / 128, M / 128), blk, 0, stream>>>(h_bf, Wfc_bf, nullptr, nullptr,
                                                           act_bf, M, F, D);
  // MLP down + residual: out = h + act @ Wfo  [8192,4096] x [4096,1024]
  gemm_bf16<3><<<dim3(D / 128, M / 128), blk, 0, stream>>>(act_bf, Wfo_bf, h_f,
                                                           (float*)d_out, nullptr, M, D, F);
}
